// AltConv_87943750353231
// MI455X (gfx1250) — compile-verified, matrix operand order checked by reading, not run on hardware
//
#include <hip/hip_runtime.h>
#include <hip/hip_bf16.h>

// ---------------------------------------------------------------------------
// AltConv (causal conv as sum of shifted GEMMs) for gfx1250 / MI455X.
// bf16 WMMA (v_wmma_f32_16x16x32_bf16), fp32 accumulate.
// Block tile 128x128, K-chunk 32, 8 wave32 waves per block.
// One x-tile of 131 rows in LDS serves all 4 taps via row offsets.
//
// MODE 2: x and weights pre-converted to bf16 in d_ws (pure-copy staging).
// MODE 1: weights bf16 in d_ws, x converted inline.
// MODE 0: fully self-contained fp32 fallback (tiny ws).
// ---------------------------------------------------------------------------

typedef __attribute__((ext_vector_type(16))) __bf16 v16bf;
typedef __attribute__((ext_vector_type(8)))  __bf16 v8bf;
typedef __attribute__((ext_vector_type(8)))  float  v8f;
typedef __attribute__((ext_vector_type(8)))  unsigned short ushort8;

#define B_   4
#define S_   8192
#define D_   1024
#define F_   1024
#define R_   4

#define TM   128              // block tile M (sequence rows)
#define TN   128              // block tile N (features)
#define TK   32               // K-chunk (matches WMMA K=32)
#define AROWS (TM + R_ - 1)   // 131 rows of x cover all 4 tap shifts
#define LDA  40               // row stride in bf16 elems (32 data + 8 pad = 80B)
#define LDB  40

__device__ __forceinline__ unsigned short f2bf(float f) {
    // round-to-nearest-even fp32 -> bf16
    unsigned int u = __float_as_uint(f);
    u += 0x7FFFu + ((u >> 16) & 1u);
    return (unsigned short)(u >> 16);
}

__device__ __forceinline__ v16bf load_frag(const __bf16* rowbase, int lhalf) {
    // WMMA 16-bit fragment: elems [lhalf..lhalf+7] and [lhalf+16..lhalf+23]
    v8bf lo = *(const v8bf*)(rowbase + lhalf);        // 16B aligned
    v8bf hi = *(const v8bf*)(rowbase + lhalf + 16);   // +32B, aligned
    return __builtin_shufflevector(lo, hi, 0,1,2,3,4,5,6,7,8,9,10,11,12,13,14,15);
}

// ---------------------------------------------------------------------------
// Prepass 1: kernels fp32 [r][d][f] -> bf16 [r][f][d] (fragment-ready).
// ---------------------------------------------------------------------------
__global__ void __launch_bounds__(256)
convert_weights(const float* __restrict__ kern, unsigned short* __restrict__ bw) {
    __shared__ __align__(16) unsigned short T[64 * 80];   // [f][d], 80-elem stride

    const int tid = threadIdx.x;
    const int d0  = blockIdx.x * 64;
    const int f0  = blockIdx.y * 64;
    const int r   = blockIdx.z;

    for (int idx = tid; idx < 64 * 16; idx += 256) {
        const int dd = idx >> 4;
        const int f4 = (idx & 15) * 4;
        const size_t g = ((size_t)(r * D_ + d0 + dd)) * F_ + f0 + f4;
        const float4 w = *(const float4*)(kern + g);
        T[(f4 + 0) * 80 + dd] = f2bf(w.x);
        T[(f4 + 1) * 80 + dd] = f2bf(w.y);
        T[(f4 + 2) * 80 + dd] = f2bf(w.z);
        T[(f4 + 3) * 80 + dd] = f2bf(w.w);
    }
    __syncthreads();

    for (int idx = tid; idx < 64 * 8; idx += 256) {
        const int ff = idx >> 3;
        const int c8 = (idx & 7) * 8;
        const ushort8 v = *(const ushort8*)(&T[ff * 80 + c8]);
        unsigned short* dst = bw + ((size_t)(r * F_ + f0 + ff)) * D_ + d0 + c8;
        *(ushort8*)dst = v;
    }
}

// ---------------------------------------------------------------------------
// Prepass 2: x fp32 -> bf16, same layout (elementwise, fully coalesced).
// ---------------------------------------------------------------------------
__global__ void __launch_bounds__(256)
convert_x(const float* __restrict__ xf, unsigned short* __restrict__ xb) {
    const size_t i = ((size_t)blockIdx.x * 256 + threadIdx.x) * 8;
    const float4 a = *(const float4*)(xf + i);
    const float4 b = *(const float4*)(xf + i + 4);
    ushort8 v = { f2bf(a.x), f2bf(a.y), f2bf(a.z), f2bf(a.w),
                  f2bf(b.x), f2bf(b.y), f2bf(b.z), f2bf(b.w) };
    *(ushort8*)(xb + i) = v;
}

// ---------------------------------------------------------------------------
// Unified WMMA GEMM, templated on staging mode.
// ---------------------------------------------------------------------------
template<int MODE>
__global__ void __launch_bounds__(256)
altconv_wmma(const float* __restrict__ xf,
             const unsigned short* __restrict__ xb,
             const float* __restrict__ kf,
             const unsigned short* __restrict__ bw,
             const float* __restrict__ biases,
             float* __restrict__ out) {
    __shared__ __align__(16) __bf16 As[AROWS * LDA];       // 131*80B = 10480B
    __shared__ __align__(16) __bf16 Bs[R_ * TN * LDB];     // 4*128*80B = 40960B

    const int tid = threadIdx.x;
    const int nb  = blockIdx.x;               // feature block (0..7)
    const int mb  = blockIdx.y;               // sequence block (0..255)

    const int g_row0 = mb * TM;               // flattened b*S+s tile base
    const int s0     = (mb % (S_ / TM)) * TM; // s base within this batch
    const int f0     = nb * TN;

    const int wave  = tid >> 5;
    const int lane  = tid & 31;
    const int wm    = (wave & 1) * 64;        // wave M offset
    const int wn    = (wave >> 1) * 32;       // wave N offset
    const int lrow  = lane & 15;
    const int lhalf = (lane >> 4) * 8;        // K half select (elem offset)
    const int mhalf = (lane >> 4) * 8;        // M offset for C/D rows

    v8f acc[4][2];
#pragma unroll
    for (int i = 0; i < 4; ++i)
#pragma unroll
        for (int j = 0; j < 2; ++j)
            acc[i][j] = (v8f)(0.0f);

    for (int kc = 0; kc < D_ / TK; ++kc) {
        const int k0 = kc * TK;

        // prefetch next K-chunk into cache hierarchy
        if constexpr (MODE >= 1) {
            if (kc + 1 < D_ / TK) {
                if constexpr (MODE == 2) {
                    const size_t gx = (size_t)(g_row0 + (tid >> 3)) * D_ + k0 + TK;
                    __builtin_prefetch(xb + gx, 0, 1);
                } else {
                    const size_t gx = (size_t)(g_row0 + (tid >> 3)) * D_ + k0 + TK;
                    __builtin_prefetch(xf + gx, 0, 1);
                }
                const size_t gb = ((size_t)(f0 + ((tid >> 1) & 127))) * D_ + k0 + TK;
                __builtin_prefetch(bw + gb, 0, 1);
            }
        }

        // ---- stage A: 131 rows x 32 elems of (shifted) x
        if constexpr (MODE == 2) {
            // pure bf16 b128 copy
            for (int idx = tid; idx < AROWS * 4; idx += 256) {   // 524 16B chunks
                const int row = idx >> 2;
                const int h   = (idx & 3) * 8;
                const int s   = s0 - (R_ - 1) + row;             // causal check
                ushort8 v = {0, 0, 0, 0, 0, 0, 0, 0};
                if (s >= 0)
                    v = *(const ushort8*)(xb + (size_t)(g_row0 - (R_ - 1) + row) * D_ + k0 + h);
                *(ushort8*)(&As[row * LDA + h]) = v;
            }
        } else {
            // fp32 load + convert
            const int col = (tid & 7) * 4;
            for (int j = (tid >> 3); j < AROWS; j += 32) {
                const int s = s0 - (R_ - 1) + j;
                float4 v = make_float4(0.f, 0.f, 0.f, 0.f);
                if (s >= 0) {
                    const size_t g = (size_t)(g_row0 - (R_ - 1) + j) * D_ + k0 + col;
                    v = *(const float4*)(xf + g);
                }
                uint2 u;
                u.x = (unsigned)f2bf(v.x) | ((unsigned)f2bf(v.y) << 16);
                u.y = (unsigned)f2bf(v.z) | ((unsigned)f2bf(v.w) << 16);
                *(uint2*)(&As[j * LDA + col]) = u;
            }
        }

        // ---- stage B: 4 taps x 128 features x 32 K of weights
        if constexpr (MODE >= 1) {
            // pure bf16 b128 copy (weights pre-transposed to [r][f][d])
            for (int idx = tid; idx < R_ * TN * 4; idx += 256) {  // 2048 16B chunks
                const int r = idx >> 9;
                const int n = (idx >> 2) & 127;
                const int h = (idx & 3) * 8;
                const v8bf w = *(const v8bf*)(bw + ((size_t)(r * F_ + f0 + n)) * D_ + k0 + h);
                *(v8bf*)(&Bs[(r * TN + n) * LDB + h]) = w;
            }
        } else {
            // fp32 load + convert + LDS transpose
#pragma unroll
            for (int r = 0; r < R_; ++r) {
                for (int idx = tid; idx < (TK * TN / 4); idx += 256) {
                    const int d  = idx >> 5;
                    const int f4 = (idx & 31) * 4;
                    const size_t g = ((size_t)(r * D_ + k0 + d)) * F_ + f0 + f4;
                    const float4 w = *(const float4*)(kf + g);
                    __bf16* brow = &Bs[(r * TN + f4) * LDB + d];
                    *(unsigned short*)(brow + 0 * LDB) = f2bf(w.x);
                    *(unsigned short*)(brow + 1 * LDB) = f2bf(w.y);
                    *(unsigned short*)(brow + 2 * LDB) = f2bf(w.z);
                    *(unsigned short*)(brow + 3 * LDB) = f2bf(w.w);
                }
            }
        }

        __syncthreads();

        // ---- compute: 4 taps x (4 M-subtiles x 2 N-subtiles) WMMAs
#pragma unroll
        for (int r = 0; r < R_; ++r) {
            const int ashift = (R_ - 1) - r;   // tap r reads shifted x rows
            v16bf afr[4];
#pragma unroll
            for (int i = 0; i < 4; ++i)
                afr[i] = load_frag(&As[(wm + i * 16 + lrow + ashift) * LDA], lhalf);
            v16bf bfr[2];
#pragma unroll
            for (int j = 0; j < 2; ++j)
                bfr[j] = load_frag(&Bs[(r * TN + wn + j * 16 + lrow) * LDB], lhalf);
#pragma unroll
            for (int i = 0; i < 4; ++i)
#pragma unroll
                for (int j = 0; j < 2; ++j)
                    acc[i][j] = __builtin_amdgcn_wmma_f32_16x16x32_bf16(
                        false, afr[i], false, bfr[j],
                        (short)0, acc[i][j], false, false);
        }

        __syncthreads();
    }

    // ---- epilogue: add summed per-tap biases, store fp32
    float biasv[2];
#pragma unroll
    for (int j = 0; j < 2; ++j) {
        const int f = f0 + wn + j * 16 + lrow;
        float b = 0.f;
#pragma unroll
        for (int r = 0; r < R_; ++r) b += biases[r * F_ + f];
        biasv[j] = b;
    }

#pragma unroll
    for (int i = 0; i < 4; ++i) {
        const int rowbase = g_row0 + wm + i * 16 + mhalf;
#pragma unroll
        for (int j = 0; j < 2; ++j) {
            const int f = f0 + wn + j * 16 + lrow;
#pragma unroll
            for (int v = 0; v < 8; ++v) {
                out[(size_t)(rowbase + v) * F_ + f] = acc[i][j][v] + biasv[j];
            }
        }
    }
}

extern "C" void kernel_launch(void* const* d_in, const int* in_sizes, int n_in,
                              void* d_out, int out_size, void* d_ws, size_t ws_size,
                              hipStream_t stream) {
    (void)in_sizes; (void)n_in; (void)out_size;
    const float* x       = (const float*)d_in[0];
    const float* kernels = (const float*)d_in[1];
    const float* biases  = (const float*)d_in[2];
    float* out = (float*)d_out;

    dim3 grid(F_ / TN, (B_ * S_) / TM);   // 8 x 256 blocks
    dim3 block(256);

    const size_t w_elems = (size_t)R_ * F_ * D_;        // 4 Mi bf16 = 8 MB
    const size_t x_elems = (size_t)B_ * S_ * D_;        // 32 Mi bf16 = 64 MB
    const size_t w_bytes = w_elems * sizeof(unsigned short);
    const size_t x_bytes = x_elems * sizeof(unsigned short);

    if (ws_size >= w_bytes + x_bytes) {
        unsigned short* bw = (unsigned short*)d_ws;
        unsigned short* xb = bw + w_elems;
        convert_weights<<<dim3(D_ / 64, F_ / 64, R_), block, 0, stream>>>(kernels, bw);
        convert_x<<<dim3((unsigned)(x_elems / (256 * 8))), block, 0, stream>>>(x, xb);
        altconv_wmma<2><<<grid, block, 0, stream>>>(nullptr, xb, nullptr, bw, biases, out);
    } else if (ws_size >= w_bytes) {
        unsigned short* bw = (unsigned short*)d_ws;
        convert_weights<<<dim3(D_ / 64, F_ / 64, R_), block, 0, stream>>>(kernels, bw);
        altconv_wmma<1><<<grid, block, 0, stream>>>(x, nullptr, nullptr, bw, biases, out);
    } else {
        altconv_wmma<0><<<grid, block, 0, stream>>>(x, nullptr, kernels, nullptr, biases, out);
    }
}
